// MixtralBlock_42949672960150
// MI455X (gfx1250) — matrix operand order repaired and checked by reading, not compiled, not run on hardware
//
#include <hip/hip_runtime.h>
#include <hip/hip_bf16.h>

// ---------------- CDNA5 WMMA helpers (bf16 in, f32 accumulate) ----------------
typedef __bf16 bf16_t;
typedef bf16_t v16bf __attribute__((ext_vector_type(16)));
typedef bf16_t v2bf  __attribute__((ext_vector_type(2)));
typedef float  v8f   __attribute__((ext_vector_type(8)));

__device__ __forceinline__ v8f wmma_bf16(v16bf a, v16bf b, v8f c) {
  return __builtin_amdgcn_wmma_f32_16x16x32_bf16(false, a, false, b, (short)0, c,
                                                 false, false);
}

// packs to one v_cvt_pk_bf16_f32 (vector-insert idiom the backend recognizes)
__device__ __forceinline__ unsigned pk_bf16(float x, float y) {
  v2bf p;
  p.x = (bf16_t)x;
  p.y = (bf16_t)y;
  return __builtin_bit_cast(unsigned, p);
}

__device__ __forceinline__ v16bf frag_from_u8(uint4 lo, uint4 hi) {
  union { unsigned u[8]; v16bf v; } x;
  x.u[0] = lo.x; x.u[1] = lo.y; x.u[2] = lo.z; x.u[3] = lo.w;
  x.u[4] = hi.x; x.u[5] = hi.y; x.u[6] = hi.z; x.u[7] = hi.w;
  return x.v;
}

// ---- LDS tile layout: bf16 pairs, rows padded to 20 dwords (80B) ------------
// A-frag (16x32): lane m=lane&15, half h: pairs j0..3 at dword 4h, j4..7 at 8+4h
__device__ __forceinline__ v16bf lds_frag_a(const unsigned* buf, int mrow) {
  int lane = threadIdx.x & 31;
  int r = lane & 15, h = lane >> 4;
  const unsigned* base = buf + (mrow + r) * 20;
  uint4 lo = *(const uint4*)(base + 4 * h);
  uint4 hi = *(const uint4*)(base + 8 + 4 * h);
  return frag_from_u8(lo, hi);
}
// B-frag (32x16): lane n=lane&15, half h: pairs j0..3 at dword 4h, j4..7 at 8+4h
__device__ __forceinline__ v16bf lds_frag_b(const unsigned* buf, int nrow) {
  int lane = threadIdx.x & 31;
  int r = lane & 15, h = lane >> 4;
  const unsigned* base = buf + (nrow + r) * 20;
  uint4 lo = *(const uint4*)(base + 4 * h);
  uint4 hi = *(const uint4*)(base + 8 + 4 * h);
  return frag_from_u8(lo, hi);
}

// ---- staging register blocks + LDS stores -----------------------------------
struct ARegs { float4 a0, a1; };          // 64x32 tile: thread t -> row t>>2, cols 8*(t&3)..+8
struct BRegs { float4 b0, b1, b2, b3; };  // 128x32 tile: thread t -> row t>>1, cols 16*(t&1)..+16

__device__ __forceinline__ ARegs gload_a(const float* p) {
  ARegs r;
  r.a0 = *(const float4*)p;
  r.a1 = *(const float4*)(p + 4);
  return r;
}
__device__ __forceinline__ BRegs gload_b(const float* p) {
  BRegs r;
  r.b0 = *(const float4*)p;
  r.b1 = *(const float4*)(p + 4);
  r.b2 = *(const float4*)(p + 8);
  r.b3 = *(const float4*)(p + 12);
  return r;
}
__device__ __forceinline__ void store_a_lds(unsigned* buf, ARegs r) {
  int t = threadIdx.x;
  unsigned* dst = buf + (t >> 2) * 20 + (t & 3) * 4;
  *(uint4*)dst = make_uint4(pk_bf16(r.a0.x, r.a0.y), pk_bf16(r.a0.z, r.a0.w),
                            pk_bf16(r.a1.x, r.a1.y), pk_bf16(r.a1.z, r.a1.w));
}
__device__ __forceinline__ void store_b_lds(unsigned* buf, BRegs r) {
  int t = threadIdx.x;
  unsigned* dst = buf + (t >> 1) * 20 + (t & 1) * 8;
  *(uint4*)dst = make_uint4(pk_bf16(r.b0.x, r.b0.y), pk_bf16(r.b0.z, r.b0.w),
                            pk_bf16(r.b1.x, r.b1.y), pk_bf16(r.b1.z, r.b1.w));
  *(uint4*)(dst + 4) =
      make_uint4(pk_bf16(r.b2.x, r.b2.y), pk_bf16(r.b2.z, r.b2.w),
                 pk_bf16(r.b3.x, r.b3.y), pk_bf16(r.b3.z, r.b3.w));
}

// ---------------- RMSNorm ----------------
__global__ __launch_bounds__(256) void rmsnorm_kernel(
    const float* __restrict__ x, const float* __restrict__ w,
    float* __restrict__ out, int H) {
  int tok = blockIdx.x;
  const float* xr = x + (size_t)tok * H;
  float ss = 0.f;
  for (int i = threadIdx.x; i < H; i += 256) {
    float v = xr[i];
    ss += v * v;
  }
#pragma unroll
  for (int m = 16; m; m >>= 1) ss += __shfl_xor(ss, m, 32);
  __shared__ float red[8];
  if ((threadIdx.x & 31) == 0) red[threadIdx.x >> 5] = ss;
  __syncthreads();
  float tot = 0.f;
#pragma unroll
  for (int i = 0; i < 8; ++i) tot += red[i];
  float rs = rsqrtf(tot / (float)H + 1e-6f);
  float* o = out + (size_t)tok * H;
  for (int i = threadIdx.x; i < H; i += 256) o[i] = w[i] * (xr[i] * rs);
}

// ---------------- Pipelined LDS GEMM: C = A[M,K] @ W[N,K]^T (+res, opt dup) --
__global__ __launch_bounds__(256) void gemm_kernel(
    const float* __restrict__ A, const float* __restrict__ W,
    const float* __restrict__ residual, float* __restrict__ C,
    float* __restrict__ C2, int M, int N, int K) {
  __shared__ unsigned Asm[2][64 * 20];
  __shared__ unsigned Bsm[2][128 * 20];
  int m0b = blockIdx.x * 64;
  int n0b = blockIdx.y * 128;
  int t = threadIdx.x;
  const float* aptr = A + (size_t)(m0b + (t >> 2)) * K + (t & 3) * 8;
  const float* bptr = W + (size_t)(n0b + (t >> 1)) * K + (t & 1) * 16;
  int wave = t >> 5;
  int wm = wave & 3, wn = wave >> 2;
  int lane = t & 31, r = lane & 15, h = lane >> 4;

  v8f acc[4] = {};
  ARegs ar = gload_a(aptr);
  BRegs br = gload_b(bptr);
  int cur = 0;
  for (int k0 = 0; k0 < K; k0 += 32) {
    store_a_lds(Asm[cur], ar);
    store_b_lds(Bsm[cur], br);
    if (k0 + 32 < K) {  // issue next K-slice loads before the barrier
      ar = gload_a(aptr + k0 + 32);
      br = gload_b(bptr + k0 + 32);
    }
    __syncthreads();
    // issue all fragment loads first, then the 4 WMMAs
    v16bf a = lds_frag_a(Asm[cur], wm * 16);
    v16bf b0 = lds_frag_b(Bsm[cur], wn * 64);
    v16bf b1 = lds_frag_b(Bsm[cur], wn * 64 + 16);
    v16bf b2 = lds_frag_b(Bsm[cur], wn * 64 + 32);
    v16bf b3 = lds_frag_b(Bsm[cur], wn * 64 + 48);
    acc[0] = wmma_bf16(a, b0, acc[0]);
    acc[1] = wmma_bf16(a, b1, acc[1]);
    acc[2] = wmma_bf16(a, b2, acc[2]);
    acc[3] = wmma_bf16(a, b3, acc[3]);
    cur ^= 1;
  }
#pragma unroll
  for (int tt = 0; tt < 4; ++tt) {
#pragma unroll
    for (int i = 0; i < 8; ++i) {
      int m = m0b + wm * 16 + (h << 3) + i;
      int n = n0b + wn * 64 + 16 * tt + r;
      size_t off = (size_t)m * N + n;
      float v = acc[tt][i];
      if (residual) v += residual[off];
      C[off] = v;
      if (C2) C2[off] = v;
    }
  }
}

// ---------------- RoPE (in place) ----------------
__global__ __launch_bounds__(256) void rope_kernel(float* __restrict__ x,
                                                   const int* __restrict__ pos,
                                                   int nheads, int HDc,
                                                   int total) {
  int idx = blockIdx.x * blockDim.x + threadIdx.x;
  if (idx >= total) return;
  int half = HDc >> 1;
  int d = idx % half;
  int t1 = idx / half;
  int head = t1 % nheads;
  int t2 = t1 / nheads;
  float p = (float)pos[t2];
  float invf = __powf(10000.0f, (-2.0f * (float)d) / (float)HDc);
  float sn, cs;
  __sincosf(p * invf, &sn, &cs);
  size_t base = ((size_t)t2 * nheads + head) * HDc;
  float x1 = x[base + d], x2 = x[base + d + half];
  x[base + d] = x1 * cs - x2 * sn;
  x[base + d + half] = x2 * cs + x1 * sn;
}

// ---------------- Flash attention: one wave per (b, head, 16-query tile) -----
#define HDIM 128
__global__ __launch_bounds__(32) void attn_kernel(
    const float* __restrict__ q, const float* __restrict__ k,
    const float* __restrict__ v, float* __restrict__ o, int S, int NQh,
    int NKVh) {
  __shared__ float Kt[32][HDIM];
  __shared__ float Vt[32][HDIM];
  __shared__ float Pt[16][34];
  int qb = blockIdx.x * 16;
  int hq = blockIdx.y;
  int b = blockIdx.z;
  int hkv = hq / (NQh / NKVh);
  int lane = threadIdx.x, r = lane & 15, hh = lane >> 4;

  const float* qrow = q + (((size_t)b * S + qb + r) * NQh + hq) * HDIM;
  v16bf qa[4];
#pragma unroll
  for (int c = 0; c < 4; ++c) {
#pragma unroll
    for (int j = 0; j < 8; ++j) {
      int kd = c * 32 + ((j >> 2) << 4) + (hh << 3) + ((j & 3) << 1);
      float2 f = *(const float2*)(qrow + kd);
      qa[c][2 * j] = (bf16_t)f.x;
      qa[c][2 * j + 1] = (bf16_t)f.y;
    }
  }

  v8f oacc[8] = {};
  float mrow[8], lrow[8];
#pragma unroll
  for (int i = 0; i < 8; ++i) {
    mrow[i] = -3.0e38f;
    lrow[i] = 0.f;
  }
  const float scale = 0.08838834764831845f;  // 1/sqrt(128)
  unsigned kt_lds = (unsigned)(size_t)&Kt[lane][0];
  unsigned vt_lds = (unsigned)(size_t)&Vt[lane][0];

  int kend = qb + 16;
  for (int kb = 0; kb < kend; kb += 32) {
    // CDNA5 async copy: each lane streams its key row straight into LDS
    {
      int key = kb + lane;
      const float* kg = k + (((size_t)b * S + key) * NKVh + hkv) * HDIM;
      const float* vg = v + (((size_t)b * S + key) * NKVh + hkv) * HDIM;
#pragma unroll
      for (int c = 0; c < 8; ++c) {
        asm volatile("global_load_async_to_lds_b128 %0, %1, off"
                     :: "v"(kt_lds + c * 64), "v"(kg + c * 16) : "memory");
        asm volatile("global_load_async_to_lds_b128 %0, %1, off"
                     :: "v"(vt_lds + c * 64), "v"(vg + c * 16) : "memory");
      }
      asm volatile("s_wait_asynccnt 0x0" ::: "memory");
    }
    __syncthreads();

    float sv[2][8];
#pragma unroll
    for (int nt = 0; nt < 2; ++nt) {
      v8f sacc = {};
#pragma unroll
      for (int c = 0; c < 4; ++c) {
        v16bf bfr;
#pragma unroll
        for (int j = 0; j < 8; ++j) {
          int kd = c * 32 + ((j >> 2) << 4) + (hh << 3) + ((j & 3) << 1);
          float2 f = *(const float2*)(&Kt[nt * 16 + r][kd]);
          bfr[2 * j] = (bf16_t)f.x;
          bfr[2 * j + 1] = (bf16_t)f.y;
        }
        sacc = wmma_bf16(qa[c], bfr, sacc);
      }
#pragma unroll
      for (int i = 0; i < 8; ++i) {
        int qm = qb + (hh << 3) + i;
        int kn = kb + nt * 16 + r;
        sv[nt][i] = (kn <= qm) ? sacc[i] * scale : -3.0e38f;
      }
    }

#pragma unroll
    for (int i = 0; i < 8; ++i) {
      float mx = fmaxf(sv[0][i], sv[1][i]);
#pragma unroll
      for (int msk = 1; msk < 16; msk <<= 1) mx = fmaxf(mx, __shfl_xor(mx, msk, 32));
      float mnew = fmaxf(mrow[i], mx);
      float alpha = __expf(mrow[i] - mnew);
      float p0 = __expf(sv[0][i] - mnew);
      float p1 = __expf(sv[1][i] - mnew);
      float ps = p0 + p1;
#pragma unroll
      for (int msk = 1; msk < 16; msk <<= 1) ps += __shfl_xor(ps, msk, 32);
      lrow[i] = lrow[i] * alpha + ps;
      mrow[i] = mnew;
#pragma unroll
      for (int tt = 0; tt < 8; ++tt) oacc[tt][i] *= alpha;
      Pt[(hh << 3) + i][r] = p0;
      Pt[(hh << 3) + i][16 + r] = p1;
    }
    __syncthreads();

    v16bf pa;
#pragma unroll
    for (int j = 0; j < 8; ++j) {
      int kk = ((j >> 2) << 4) + (hh << 3) + ((j & 3) << 1);
      float2 f = *(const float2*)(&Pt[r][kk]);
      pa[2 * j] = (bf16_t)f.x;
      pa[2 * j + 1] = (bf16_t)f.y;
    }
#pragma unroll
    for (int tt = 0; tt < 8; ++tt) {
      v16bf bv;
#pragma unroll
      for (int j = 0; j < 8; ++j) {
        int key = ((j >> 2) << 4) + (hh << 3) + ((j & 3) << 1);
        bv[2 * j] = (bf16_t)Vt[key][tt * 16 + r];
        bv[2 * j + 1] = (bf16_t)Vt[key + 1][tt * 16 + r];
      }
      oacc[tt] = wmma_bf16(pa, bv, oacc[tt]);
    }
    __syncthreads();
  }

#pragma unroll
  for (int i = 0; i < 8; ++i) {
    float inv = 1.0f / lrow[i];
    int m = qb + (hh << 3) + i;
    float* orow = o + ((size_t)b * S + m) * (NQh * HDIM) + (size_t)hq * HDIM;
#pragma unroll
    for (int tt = 0; tt < 8; ++tt) orow[tt * 16 + r] = oacc[tt][i] * inv;
  }
}

// ---------------- Router ----------------
__global__ __launch_bounds__(256) void router_kernel(
    const float* __restrict__ x, const float* __restrict__ gw,
    int* __restrict__ eidx, float* __restrict__ ewgt, int* __restrict__ ecnt,
    int H, int NT) {
  int tok = blockIdx.x;
  int wave = threadIdx.x >> 5, lane = threadIdx.x & 31;
  __shared__ float logits[8];
  const float* xr = x + (size_t)tok * H;
  const float* wr = gw + (size_t)wave * H;
  float s = 0.f;
  for (int i = lane; i < H; i += 32) s += xr[i] * wr[i];
#pragma unroll
  for (int m = 16; m; m >>= 1) s += __shfl_xor(s, m, 32);
  if (lane == 0) logits[wave] = s;
  __syncthreads();
  if (threadIdx.x == 0) {
    float mx = logits[0];
#pragma unroll
    for (int e = 1; e < 8; ++e) mx = fmaxf(mx, logits[e]);
    float p[8], se = 0.f;
#pragma unroll
    for (int e = 0; e < 8; ++e) {
      p[e] = __expf(logits[e] - mx);
      se += p[e];
    }
#pragma unroll
    for (int e = 0; e < 8; ++e) p[e] /= se;
    int i0 = 0;
#pragma unroll
    for (int e = 1; e < 8; ++e)
      if (p[e] > p[i0]) i0 = e;
    int i1 = (i0 == 0) ? 1 : 0;
#pragma unroll
    for (int e = 0; e < 8; ++e)
      if (e != i0 && p[e] > p[i1]) i1 = e;
    float ws = p[i0] + p[i1];
    int s0 = atomicAdd(&ecnt[i0], 1);
    eidx[i0 * NT + s0] = tok;
    ewgt[i0 * NT + s0] = p[i0] / ws;
    int s1 = atomicAdd(&ecnt[i1], 1);
    eidx[i1 * NT + s1] = tok;
    ewgt[i1 * NT + s1] = p[i1] / ws;
  }
}

__global__ void zero_int_kernel(int* p, int n) {
  int i = blockIdx.x * blockDim.x + threadIdx.x;
  if (i < n) p[i] = 0;
}

// ---------------- MoE GEMM1: h1 = silu(x@w1^T)*(x@w3^T), gathered rows -------
__global__ __launch_bounds__(256) void moe_gemm1_kernel(
    const float* __restrict__ x, const float* __restrict__ w1e,
    const float* __restrict__ w3e, float* __restrict__ h1,
    const int* __restrict__ eidx, const int* __restrict__ cnt_ptr, int H,
    int INTERn) {
  __shared__ unsigned Asm[2][64 * 20];
  __shared__ unsigned B1s[2][128 * 20];
  __shared__ unsigned B3s[2][128 * 20];
  int cnt = *cnt_ptr;
  int m0b = blockIdx.x * 64;
  if (m0b >= cnt) return;  // uniform over block: no barrier divergence
  int n0b = blockIdx.y * 128;
  int t = threadIdx.x;
  int arow = m0b + (t >> 2);
  int tok = eidx[arow < cnt ? arow : 0];
  const float* aptr = x + (size_t)tok * H + (t & 3) * 8;
  const float* b1p = w1e + (size_t)(n0b + (t >> 1)) * H + (t & 1) * 16;
  const float* b3p = w3e + (size_t)(n0b + (t >> 1)) * H + (t & 1) * 16;
  int wave = t >> 5;
  int wm = wave & 3, wn = wave >> 2;
  int lane = t & 31, r = lane & 15, h = lane >> 4;

  v8f accg[4] = {}, accu[4] = {};
  ARegs ar = gload_a(aptr);
  BRegs r1 = gload_b(b1p);
  BRegs r3 = gload_b(b3p);
  int cur = 0;
  for (int k0 = 0; k0 < H; k0 += 32) {
    store_a_lds(Asm[cur], ar);
    store_b_lds(B1s[cur], r1);
    store_b_lds(B3s[cur], r3);
    if (k0 + 32 < H) {
      ar = gload_a(aptr + k0 + 32);
      r1 = gload_b(b1p + k0 + 32);
      r3 = gload_b(b3p + k0 + 32);
    }
    __syncthreads();
    v16bf a = lds_frag_a(Asm[cur], wm * 16);
    {
      v16bf b0 = lds_frag_b(B1s[cur], wn * 64);
      v16bf b1 = lds_frag_b(B1s[cur], wn * 64 + 16);
      v16bf b2 = lds_frag_b(B1s[cur], wn * 64 + 32);
      v16bf b3 = lds_frag_b(B1s[cur], wn * 64 + 48);
      accg[0] = wmma_bf16(a, b0, accg[0]);
      accg[1] = wmma_bf16(a, b1, accg[1]);
      accg[2] = wmma_bf16(a, b2, accg[2]);
      accg[3] = wmma_bf16(a, b3, accg[3]);
    }
    {
      v16bf b0 = lds_frag_b(B3s[cur], wn * 64);
      v16bf b1 = lds_frag_b(B3s[cur], wn * 64 + 16);
      v16bf b2 = lds_frag_b(B3s[cur], wn * 64 + 32);
      v16bf b3 = lds_frag_b(B3s[cur], wn * 64 + 48);
      accu[0] = wmma_bf16(a, b0, accu[0]);
      accu[1] = wmma_bf16(a, b1, accu[1]);
      accu[2] = wmma_bf16(a, b2, accu[2]);
      accu[3] = wmma_bf16(a, b3, accu[3]);
    }
    cur ^= 1;
  }
#pragma unroll
  for (int tt = 0; tt < 4; ++tt) {
#pragma unroll
    for (int i = 0; i < 8; ++i) {
      int m = m0b + wm * 16 + (h << 3) + i;
      if (m < cnt) {
        float g = accg[tt][i], u = accu[tt][i];
        float sig = 1.f / (1.f + __expf(-g));
        h1[(size_t)m * INTERn + n0b + wn * 64 + 16 * tt + r] = g * sig * u;
      }
    }
  }
}

// ---------------- MoE GEMM2: out[tok] += w * (h1 @ w2^T) ---------------------
__global__ __launch_bounds__(256) void moe_gemm2_kernel(
    const float* __restrict__ h1, const float* __restrict__ w2e,
    float* __restrict__ out, const int* __restrict__ eidx,
    const float* __restrict__ ewgt, const int* __restrict__ cnt_ptr, int INTERn,
    int H) {
  __shared__ unsigned Asm[2][64 * 20];
  __shared__ unsigned Bsm[2][128 * 20];
  int cnt = *cnt_ptr;
  int m0b = blockIdx.x * 64;
  if (m0b >= cnt) return;
  int n0b = blockIdx.y * 128;
  int t = threadIdx.x;
  const float* aptr = h1 + (size_t)(m0b + (t >> 2)) * INTERn + (t & 3) * 8;
  const float* bptr = w2e + (size_t)(n0b + (t >> 1)) * INTERn + (t & 1) * 16;
  int wave = t >> 5;
  int wm = wave & 3, wn = wave >> 2;
  int lane = t & 31, r = lane & 15, h = lane >> 4;

  v8f acc[4] = {};
  ARegs ar = gload_a(aptr);
  BRegs br = gload_b(bptr);
  int cur = 0;
  for (int k0 = 0; k0 < INTERn; k0 += 32) {
    store_a_lds(Asm[cur], ar);
    store_b_lds(Bsm[cur], br);
    if (k0 + 32 < INTERn) {
      ar = gload_a(aptr + k0 + 32);
      br = gload_b(bptr + k0 + 32);
    }
    __syncthreads();
    v16bf a = lds_frag_a(Asm[cur], wm * 16);
    v16bf b0 = lds_frag_b(Bsm[cur], wn * 64);
    v16bf b1 = lds_frag_b(Bsm[cur], wn * 64 + 16);
    v16bf b2 = lds_frag_b(Bsm[cur], wn * 64 + 32);
    v16bf b3 = lds_frag_b(Bsm[cur], wn * 64 + 48);
    acc[0] = wmma_bf16(a, b0, acc[0]);
    acc[1] = wmma_bf16(a, b1, acc[1]);
    acc[2] = wmma_bf16(a, b2, acc[2]);
    acc[3] = wmma_bf16(a, b3, acc[3]);
    cur ^= 1;
  }
#pragma unroll
  for (int i = 0; i < 8; ++i) {
    int m = m0b + wm * 16 + (h << 3) + i;
    if (m < cnt) {
      int tok = eidx[m];
      float cw = ewgt[m];
      float* orow = out + (size_t)tok * H;
#pragma unroll
      for (int tt = 0; tt < 4; ++tt) {
        int n = n0b + wn * 64 + 16 * tt + r;
        orow[n] += cw * acc[(unsigned)tt][i];
      }
    }
  }
}

// ---------------- Host orchestration ----------------
extern "C" void kernel_launch(void* const* d_in, const int* in_sizes, int n_in,
                              void* d_out, int out_size, void* d_ws,
                              size_t ws_size, hipStream_t stream) {
  (void)in_sizes; (void)n_in; (void)out_size; (void)ws_size;
  constexpr int B = 2, S = 1024, H = 2048;
  constexpr int NQ = 16, NKV = 8, HD = 128;
  constexpr int E = 8, INTER = 4096;
  constexpr int NT = B * S;

  const float* hs   = (const float*)d_in[0];
  const int*   pos  = (const int*)d_in[1];
  const float* ln1w = (const float*)d_in[2];
  const float* qw   = (const float*)d_in[3];
  const float* kw   = (const float*)d_in[4];
  const float* vw   = (const float*)d_in[5];
  const float* ow   = (const float*)d_in[6];
  const float* ln2w = (const float*)d_in[7];
  const float* gw   = (const float*)d_in[8];
  const float* w1   = (const float*)d_in[9];
  const float* w2   = (const float*)d_in[10];
  const float* w3   = (const float*)d_in[11];
  float* out = (float*)d_out;

  char* wsp = (char*)d_ws;
  auto alloc = [&](size_t bytes) {
    char* p = wsp;
    wsp += (bytes + 255) & ~(size_t)255;
    return p;
  };
  float* hsn1   = (float*)alloc(sizeof(float) * (size_t)NT * H);
  float* qbuf   = (float*)alloc(sizeof(float) * (size_t)NT * NQ * HD);
  float* kbuf   = (float*)alloc(sizeof(float) * (size_t)NT * NKV * HD);
  float* vbuf   = (float*)alloc(sizeof(float) * (size_t)NT * NKV * HD);
  float* attn   = (float*)alloc(sizeof(float) * (size_t)NT * H);
  float* hidden = (float*)alloc(sizeof(float) * (size_t)NT * H);
  float* hs2    = (float*)alloc(sizeof(float) * (size_t)NT * H);
  float* h1     = (float*)alloc(sizeof(float) * (size_t)NT * INTER);
  int*   eidx   = (int*)alloc(sizeof(int) * (size_t)E * NT);
  float* ewgt   = (float*)alloc(sizeof(float) * (size_t)E * NT);
  int*   ecnt   = (int*)alloc(sizeof(int) * E);

  rmsnorm_kernel<<<NT, 256, 0, stream>>>(hs, ln1w, hsn1, H);
  gemm_kernel<<<dim3(NT / 64, (NQ * HD) / 128), 256, 0, stream>>>(
      hsn1, qw, nullptr, qbuf, nullptr, NT, NQ * HD, H);
  gemm_kernel<<<dim3(NT / 64, (NKV * HD) / 128), 256, 0, stream>>>(
      hsn1, kw, nullptr, kbuf, nullptr, NT, NKV * HD, H);
  gemm_kernel<<<dim3(NT / 64, (NKV * HD) / 128), 256, 0, stream>>>(
      hsn1, vw, nullptr, vbuf, nullptr, NT, NKV * HD, H);
  {
    int tq = NT * NQ * (HD / 2);
    rope_kernel<<<(tq + 255) / 256, 256, 0, stream>>>(qbuf, pos, NQ, HD, tq);
    int tk = NT * NKV * (HD / 2);
    rope_kernel<<<(tk + 255) / 256, 256, 0, stream>>>(kbuf, pos, NKV, HD, tk);
  }
  attn_kernel<<<dim3(S / 16, NQ, B), 32, 0, stream>>>(qbuf, kbuf, vbuf, attn, S,
                                                      NQ, NKV);
  gemm_kernel<<<dim3(NT / 64, H / 128), 256, 0, stream>>>(attn, ow, hs, hidden,
                                                          out, NT, H, H);
  rmsnorm_kernel<<<NT, 256, 0, stream>>>(hidden, ln2w, hs2, H);
  zero_int_kernel<<<1, 32, 0, stream>>>(ecnt, E);
  router_kernel<<<NT, 256, 0, stream>>>(hs2, gw, eidx, ewgt, ecnt, H, NT);
  for (int e = 0; e < E; ++e) {
    const float* w1e = w1 + (size_t)e * INTER * H;
    const float* w3e = w3 + (size_t)e * INTER * H;
    const float* w2e = w2 + (size_t)e * H * INTER;
    moe_gemm1_kernel<<<dim3(NT / 64, INTER / 128), 256, 0, stream>>>(
        hs2, w1e, w3e, h1, eidx + (size_t)e * NT, ecnt + e, H, INTER);
    moe_gemm2_kernel<<<dim3(NT / 64, H / 128), 256, 0, stream>>>(
        h1, w2e, out, eidx + (size_t)e * NT, ewgt + (size_t)e * NT, ecnt + e,
        INTER, H);
  }
}
